// EGraphSAGE_54176717471771
// MI455X (gfx1250) — matrix operand order repaired and checked by reading, not compiled, not run on hardware
//
#include <hip/hip_runtime.h>

// ---------------------------------------------------------------------------
// EGraphSAGE (2 layers) for MI455X / gfx1250, wave32, bf16 WMMA w/ f32 accum.
//
//   N = 50000 nodes, E = 800000 edges, all feature dims = 128.
//   Node GEMM: [N,384] x [384,128]^T ; Edge GEMM: [E,256] x [256,128]^T.
//
// v3: - h / nf pre-converted to bf16 once per layer -> edge-MLP staging is
//       pure uint4 gather->LDS copies (no per-edge convert VALU, half the
//       L2 gather bytes).
//     - A-fragment double buffering in the WMMA k-loop (overlap ds_load
//       latency with matrix ops).
//     - B fragments register-resident, reused across 4 M-tiles per block.
// ---------------------------------------------------------------------------

#define NNODES 50000
#define NEDGES 800000
#define DIM    128
#define KNODE  384   // [nf | h_neigh(256)]
#define KEDGE  256   // [h[u] | h[v]]
#define MT     4     // M-tiles (of 16 rows) per block

typedef __attribute__((ext_vector_type(16))) __bf16 v16bf;
typedef __attribute__((ext_vector_type(8)))  float  v8f;

union Frag32B {             // one WMMA bf16 A/B fragment: 16 bf16 = 32 bytes
  v16bf v;
  uint4 q[2];
};

// round-to-nearest-even f32 -> bf16 (bit pattern in low 16)
__device__ __forceinline__ unsigned int f2bf(float f) {
  unsigned int x = __float_as_uint(f);
  x += 0x7FFFu + ((x >> 16) & 1u);
  return x >> 16;
}

// pack 8 f32 -> 8 bf16 in a uint4 (16 bytes)
__device__ __forceinline__ uint4 pack8(float4 a, float4 b) {
  uint4 r;
  r.x = f2bf(a.x) | (f2bf(a.y) << 16);
  r.y = f2bf(a.z) | (f2bf(a.w) << 16);
  r.z = f2bf(b.x) | (f2bf(b.y) << 16);
  r.w = f2bf(b.z) | (f2bf(b.w) << 16);
  return r;
}

// ---------------------------------------------------------------------------
// Utility kernels
// ---------------------------------------------------------------------------
__global__ void zero_f32(float* __restrict__ p, int n) {
  int i = blockIdx.x * blockDim.x + threadIdx.x;
  if (i < n) p[i] = 0.0f;
}

__global__ void convert_f32_bf16(const float* __restrict__ src,
                                 unsigned short* __restrict__ dst, int n) {
  int i = blockIdx.x * blockDim.x + threadIdx.x;
  if (i < n) dst[i] = (unsigned short)f2bf(src[i]);
}

// ---------------------------------------------------------------------------
// Scatter-sum of per-edge message [nf[u] | ef] onto destination nodes.
// 8 waves per block, one edge per wave; f32 atomics resolve in L2.
// ---------------------------------------------------------------------------
__global__ __launch_bounds__(256) void scatter_edges(
    const float* __restrict__ nf,    // [N,128] (node features / h)
    const float* __restrict__ ef,    // [E,128] (edge features / e)
    const int*   __restrict__ u,
    const int*   __restrict__ v,
    float* __restrict__ sums,        // [N,256]
    float* __restrict__ cnt)         // [N]
{
  int wave = threadIdx.x >> 5;
  int lane = threadIdx.x & 31;
  long long e = (long long)blockIdx.x * 8 + wave;
  if (e >= NEDGES) return;
  int us = u[e];
  int vs = v[e];
  const float4* srow = (const float4*)(nf + (long long)us * DIM);
  const float4* erow = (const float4*)(ef + e * (long long)DIM);
  float*        drow = sums + (long long)vs * (2 * DIM);

  float4 sv = srow[lane];            // cols 4*lane .. 4*lane+3
  float4 ev = erow[lane];
  int c = lane * 4;
  atomicAdd(&drow[c + 0], sv.x);
  atomicAdd(&drow[c + 1], sv.y);
  atomicAdd(&drow[c + 2], sv.z);
  atomicAdd(&drow[c + 3], sv.w);
  atomicAdd(&drow[DIM + c + 0], ev.x);
  atomicAdd(&drow[DIM + c + 1], ev.y);
  atomicAdd(&drow[DIM + c + 2], ev.z);
  atomicAdd(&drow[DIM + c + 3], ev.w);
  if (lane == 0) atomicAdd(&cnt[vs], 1.0f);
}

// ---------------------------------------------------------------------------
// Double-buffered WMMA sweep over one staged A-tile (J k-steps of 32).
// ---------------------------------------------------------------------------
template <int J>
__device__ __forceinline__ v8f wmma_sweep(const unsigned short* __restrict__ arow,
                                          const Frag32B* __restrict__ bfr,
                                          int hi) {
  v8f acc = {};
  Frag32B a0, a1;
  a0.q[0] = *(const uint4*)&arow[hi * 8];
  a0.q[1] = *(const uint4*)&arow[16 + hi * 8];
#pragma unroll
  for (int j = 0; j < J; j += 2) {
    a1.q[0] = *(const uint4*)&arow[(j + 1) * 32 + hi * 8];
    a1.q[1] = *(const uint4*)&arow[(j + 1) * 32 + 16 + hi * 8];
    acc = __builtin_amdgcn_wmma_f32_16x16x32_bf16(
        false, a0.v, false, bfr[j].v, (short)0, acc, false, false);
    if (j + 2 < J) {
      a0.q[0] = *(const uint4*)&arow[(j + 2) * 32 + hi * 8];
      a0.q[1] = *(const uint4*)&arow[(j + 2) * 32 + 16 + hi * 8];
    }
    acc = __builtin_amdgcn_wmma_f32_16x16x32_bf16(
        false, a1.v, false, bfr[j + 1].v, (short)0, acc, false, false);
  }
  return acc;
}

// ---------------------------------------------------------------------------
// Node MLP: h_out = relu([x | sums/max(cnt,1)] @ W^T + b)
// Block = 256 threads (8 waves) = 64 output rows (4 tiles), 128 output cols.
// x is pre-converted bf16; sums/cnt half converts during staging (needs the
// divide anyway). B fragments register-resident across all 4 M-tiles.
// ---------------------------------------------------------------------------
__global__ __launch_bounds__(256) void node_mlp(
    const unsigned short* __restrict__ x_bf, // [N,128] bf16
    const float* __restrict__ sums,          // [N,256]
    const float* __restrict__ cnt,           // [N]
    const unsigned short* __restrict__ Wbf,  // [128,384] bf16 (row = out col)
    const float* __restrict__ bias,          // [128]
    float* __restrict__ hout)                // [N,128] f32
{
  __shared__ __align__(16) unsigned short ldsA[16 * KNODE];

  int lane  = threadIdx.x & 31;
  int tileN = (threadIdx.x >> 5) * 16;
  int nn = lane & 15;
  int hi = lane >> 4;

  // Preload B fragments (W is L2-resident bf16): B[k][n] = W[n][k].
  Frag32B bfr[KNODE / 32];
  const unsigned short* brow = &Wbf[(long long)(tileN + nn) * KNODE];
#pragma unroll
  for (int j = 0; j < KNODE / 32; ++j) {
    bfr[j].q[0] = *(const uint4*)&brow[j * 32 + hi * 8];
    bfr[j].q[1] = *(const uint4*)&brow[j * 32 + 16 + hi * 8];
  }
  float bval = bias[tileN + nn];

  for (int t = 0; t < MT; ++t) {
    int tileM = (blockIdx.x * MT + t) * 16;
    if (tileM >= NNODES) break;              // uniform across block
    if (t) __syncthreads();                  // prior tile's readers done

    // Stage A tile: 16 x 384 = 6144 elems, 8 per store, 3 stores per thread.
#pragma unroll
    for (int i = 0; i < 3; ++i) {
      int base = (threadIdx.x + i * 256) * 8;
      int m = base / KNODE;
      int k = base - m * KNODE;
      int row = tileM + m;
      if (k < DIM) {                         // plain bf16 copy
        *(uint4*)&ldsA[base] =
            *(const uint4*)&x_bf[(long long)row * DIM + k];
      } else {                               // mean + convert
        float inv = 1.0f / fmaxf(cnt[row], 1.0f);
        const float4* p =
            (const float4*)&sums[(long long)row * (2 * DIM) + (k - DIM)];
        float4 x0 = p[0], x1 = p[1];
        x0.x *= inv; x0.y *= inv; x0.z *= inv; x0.w *= inv;
        x1.x *= inv; x1.y *= inv; x1.z *= inv; x1.w *= inv;
        *(uint4*)&ldsA[base] = pack8(x0, x1);
      }
    }
    __syncthreads();

    v8f acc = wmma_sweep<KNODE / 32>(&ldsA[(lane & 15) * KNODE], bfr, hi);

#pragma unroll
    for (int r = 0; r < 8; ++r) {            // C/D: row = r + 8*hi, col = nn
      int m = tileM + r + hi * 8;
      float o = acc[r] + bval;
      hout[(long long)m * DIM + tileN + nn] = fmaxf(o, 0.0f);
    }
  }
}

// ---------------------------------------------------------------------------
// Edge MLP: e_out = relu([h[u] | h[v]] @ W^T + b)
// h already bf16 -> staging is pure uint4 gather->LDS copies (no convert),
// and gathers move 2 B/elem out of L2 instead of 4 B.
// ---------------------------------------------------------------------------
__global__ __launch_bounds__(256) void edge_mlp(
    const unsigned short* __restrict__ h_bf, // [N,128] bf16
    const int*   __restrict__ u,
    const int*   __restrict__ v,
    const unsigned short* __restrict__ Wbf,  // [128,256] bf16
    const float* __restrict__ bias,          // [128]
    float* __restrict__ eout)                // [E,128] f32
{
  __shared__ __align__(16) unsigned short ldsA[16 * KEDGE];
  __shared__ int uv[32];

  int lane  = threadIdx.x & 31;
  int tileN = (threadIdx.x >> 5) * 16;
  int nn = lane & 15;
  int hi = lane >> 4;

  Frag32B bfr[KEDGE / 32];
  const unsigned short* brow = &Wbf[(long long)(tileN + nn) * KEDGE];
#pragma unroll
  for (int j = 0; j < KEDGE / 32; ++j) {
    bfr[j].q[0] = *(const uint4*)&brow[j * 32 + hi * 8];
    bfr[j].q[1] = *(const uint4*)&brow[j * 32 + 16 + hi * 8];
  }
  float bval = bias[tileN + nn];

  for (int t = 0; t < MT; ++t) {
    long long tileM = ((long long)blockIdx.x * MT + t) * 16;
    if (t) __syncthreads();

    if (threadIdx.x < 16)      uv[threadIdx.x] = u[tileM + threadIdx.x];
    else if (threadIdx.x < 32) uv[threadIdx.x] = v[tileM + threadIdx.x - 16];
    __syncthreads();

    // Stage A tile: 16 edges x 256 elems = 512 uint4 copies, 2 per thread.
#pragma unroll
    for (int i = 0; i < 2; ++i) {
      int base = (threadIdx.x + i * 256) * 8;
      int m = base >> 8;                     // KEDGE == 256
      int k = base & 255;
      int src = (k < DIM) ? uv[m] : uv[16 + m];
      *(uint4*)&ldsA[base] =
          *(const uint4*)&h_bf[(long long)src * DIM + (k & (DIM - 1))];
    }
    __syncthreads();

    v8f acc = wmma_sweep<KEDGE / 32>(&ldsA[(lane & 15) * KEDGE], bfr, hi);

#pragma unroll
    for (int r = 0; r < 8; ++r) {
      long long m = tileM + r + hi * 8;
      float o = acc[r] + bval;
      eout[m * DIM + tileN + nn] = fmaxf(o, 0.0f);
    }
  }
}

// ---------------------------------------------------------------------------
// Launch
// ---------------------------------------------------------------------------
extern "C" void kernel_launch(void* const* d_in, const int* in_sizes, int n_in,
                              void* d_out, int out_size, void* d_ws, size_t ws_size,
                              hipStream_t stream) {
  (void)in_sizes; (void)n_in; (void)out_size; (void)ws_size;

  const float* nf  = (const float*)d_in[0];
  const float* ef  = (const float*)d_in[1];
  const int*   u   = (const int*)  d_in[2];
  const int*   v   = (const int*)  d_in[3];
  const float* W1a = (const float*)d_in[4];
  const float* b1a = (const float*)d_in[5];
  const float* W1e = (const float*)d_in[6];
  const float* b1e = (const float*)d_in[7];
  const float* W2a = (const float*)d_in[8];
  const float* b2a = (const float*)d_in[9];
  const float* W2e = (const float*)d_in[10];
  const float* b2e = (const float*)d_in[11];

  // Workspace layout
  char* ws = (char*)d_ws;
  float* sums = (float*)ws;  ws += sizeof(float) * (size_t)NNODES * 256;
  float* cnt  = (float*)ws;  ws += sizeof(float) * (size_t)NNODES;
  float* h1   = (float*)ws;  ws += sizeof(float) * (size_t)NNODES * DIM;
  float* e1   = (float*)ws;  ws += sizeof(float) * (size_t)NEDGES * DIM;
  unsigned short* nf_bf = (unsigned short*)ws; ws += 2u * (size_t)NNODES * DIM;
  unsigned short* h1_bf = (unsigned short*)ws; ws += 2u * (size_t)NNODES * DIM;
  unsigned short* h2_bf = (unsigned short*)ws; ws += 2u * (size_t)NNODES * DIM;
  unsigned short* W1a_bf = (unsigned short*)ws; ws += 2u * DIM * KNODE;
  unsigned short* W1e_bf = (unsigned short*)ws; ws += 2u * DIM * KEDGE;
  unsigned short* W2a_bf = (unsigned short*)ws; ws += 2u * DIM * KNODE;
  unsigned short* W2e_bf = (unsigned short*)ws; ws += 2u * DIM * KEDGE;

  float* h2 = (float*)d_out;                           // [N,128]
  float* e2 = (float*)d_out + (size_t)NNODES * DIM;    // [E,128]

  const int nFeat = NNODES * DIM;

  // bf16 conversions: weights (tiny) + node features.
  convert_f32_bf16<<<(DIM * KNODE + 255) / 256, 256, 0, stream>>>(W1a, W1a_bf, DIM * KNODE);
  convert_f32_bf16<<<(DIM * KEDGE + 255) / 256, 256, 0, stream>>>(W1e, W1e_bf, DIM * KEDGE);
  convert_f32_bf16<<<(DIM * KNODE + 255) / 256, 256, 0, stream>>>(W2a, W2a_bf, DIM * KNODE);
  convert_f32_bf16<<<(DIM * KEDGE + 255) / 256, 256, 0, stream>>>(W2e, W2e_bf, DIM * KEDGE);
  convert_f32_bf16<<<(nFeat + 255) / 256, 256, 0, stream>>>(nf, nf_bf, nFeat);

  const int nAgg = NNODES * 256 + NNODES;  // sums + cnt are contiguous
  const int nodeBlocks = (NNODES / 16 + MT - 1) / MT;  // 782 (last: 1 tile)
  const int edgeBlocks = NEDGES / (16 * MT);           // 12500 exact

  // ---------------- Layer 1 ----------------
  zero_f32<<<(nAgg + 255) / 256, 256, 0, stream>>>(sums, nAgg);
  scatter_edges<<<NEDGES / 8, 256, 0, stream>>>(nf, ef, u, v, sums, cnt);
  node_mlp<<<nodeBlocks, 256, 0, stream>>>(nf_bf, sums, cnt, W1a_bf, b1a, h1);
  convert_f32_bf16<<<(nFeat + 255) / 256, 256, 0, stream>>>(h1, h1_bf, nFeat);
  edge_mlp<<<edgeBlocks, 256, 0, stream>>>(h1_bf, u, v, W1e_bf, b1e, e1);

  // ---------------- Layer 2 ----------------
  zero_f32<<<(nAgg + 255) / 256, 256, 0, stream>>>(sums, nAgg);
  scatter_edges<<<NEDGES / 8, 256, 0, stream>>>(h1, e1, u, v, sums, cnt);
  node_mlp<<<nodeBlocks, 256, 0, stream>>>(h1_bf, sums, cnt, W2a_bf, b2a, h2);
  convert_f32_bf16<<<(nFeat + 255) / 256, 256, 0, stream>>>(h2, h2_bf, nFeat);
  edge_mlp<<<edgeBlocks, 256, 0, stream>>>(h2_bf, u, v, W2e_bf, b2e, e2);
}
